// NovaWinogradConv2d_9431748182311
// MI455X (gfx1250) — matrix operand order repaired and checked by reading, not compiled, not run on hardware
//
#include <hip/hip_runtime.h>

typedef _Float16 v8h  __attribute__((ext_vector_type(8)));
typedef _Float16 v16h __attribute__((ext_vector_type(16)));
typedef float    v8f  __attribute__((ext_vector_type(8)));

#define BATCH 32
#define CIN   256
#define COUT  256
#define HH    112
#define WW    112
#define HP    114                // padded (halo) height
#define WP    114                // padded (halo) width
#define HWSZ  (HH * WW)          // 12544
#define NPIX  (BATCH * HWSZ)     // 401408

#define BM 128                   // cout tile per block
#define BN 128                   // pixel tile per block
#define BK 32                    // K slice per WMMA step
#define LDSS 40                  // padded LDS row stride in halves (80B = 5*16B)
#define KSTEPS 72                // 9 taps * 8 cin-blocks

union H16 { v16h v; v8h h[2]; };

// ---------------------------------------------------------------------------
// Prep 1: weight[cout][cin][3][3] fp32  ->  wt[tap][cout][cin] fp16
// ---------------------------------------------------------------------------
__global__ void nova_wt_xform(const float* __restrict__ w, _Float16* __restrict__ wt) {
    int idx = blockIdx.x * 256 + threadIdx.x;        // 9*COUT*CIN total
    int tap = idx / (COUT * CIN);
    int rem = idx - tap * (COUT * CIN);
    int co  = rem / CIN;
    int ci  = rem - co * CIN;
    wt[idx] = (_Float16)w[(co * CIN + ci) * 9 + tap];
}

// ---------------------------------------------------------------------------
// Prep 2: x NCHW fp16 -> xt NHWC fp16 with zero halo: xt[b][h+1][w+1][ci]
// (border pre-zeroed by hipMemsetAsync). LDS-tiled 32x32 transpose.
// grid: (w-tiles=4, ci-tiles=8, b*h=3584), block (32,8)
// ---------------------------------------------------------------------------
__global__ void nova_nchw_to_nhwc(const _Float16* __restrict__ x, _Float16* __restrict__ xt) {
    __shared__ _Float16 tile[32][33];
    const int bh  = blockIdx.z;
    const int b   = bh / HH;
    const int h   = bh - b * HH;
    const int w0  = blockIdx.x * 32;
    const int ci0 = blockIdx.y * 32;
    const int tx  = threadIdx.x;   // 0..31
    const int ty  = threadIdx.y;   // 0..7

    #pragma unroll
    for (int j = ty; j < 32; j += 8) {
        int wcol = w0 + tx;
        _Float16 v = (_Float16)0.0f;
        if (wcol < WW) v = x[((size_t)(b * CIN + ci0 + j) * HH + h) * WW + wcol];
        tile[j][tx] = v;
    }
    __syncthreads();
    #pragma unroll
    for (int j = ty; j < 32; j += 8) {
        int wcol = w0 + j;
        if (wcol < WW)
            xt[((size_t)(b * HP + (h + 1)) * WP + (wcol + 1)) * CIN + ci0 + tx] = tile[tx][j];
    }
}

// ---------------------------------------------------------------------------
// Main: implicit-GEMM conv with double-buffered async-to-LDS staging.
// C[cout][pix] = sum_{tap,cin} Wt[tap][cout][cin] * Xt[pix@tap][cin]
// accumulated with v_wmma_f32_16x16x32_f16.
// grid (NPIX/BN, COUT/BM), block 256 (8 wave32)
// ---------------------------------------------------------------------------
__global__ __launch_bounds__(256, 2)
void nova_conv_wmma(const _Float16* __restrict__ xt,
                    const _Float16* __restrict__ wt,
                    _Float16* __restrict__ out) {
    __shared__ _Float16 As[2][BM * LDSS];   // [buf][cout][k]
    __shared__ _Float16 Bs[2][BN * LDSS];   // [buf][pixel][k]

    const int tid   = threadIdx.x;
    const int lane  = tid & 31;
    const int wv    = tid >> 5;
    const int pix0  = blockIdx.x * BN;
    const int cout0 = blockIdx.y * BM;

    // staging role: each thread fills 2 rows x 8 halves of each tile
    const int kc8    = (tid & 3) * 8;
    const int rowbse = tid >> 2;            // 0..63 (+64 for second row)
    int pb[2], poh[2], pow_[2];
    #pragma unroll
    for (int t = 0; t < 2; ++t) {
        int row = rowbse + t * 64;
        int p   = pix0 + row;
        int b   = p / HWSZ;
        int r   = p - b * HWSZ;
        pb[t] = b; poh[t] = r / WW; pow_[t] = r - (r / WW) * WW;
    }

    const int m_wave = (wv & 1) * 64;
    const int n_wave = (wv >> 1) * 32;
    const int half   = (lane < 16) ? 0 : 1;
    const int l16    = lane & 15;

    // issue async copies for K-step s into LDS buffer buf
    auto stage = [&](int buf, int s) {
        const int tap = s >> 3;
        const int cb  = s & 7;
        const int dr  = tap / 3;            // 0..2 (halo-shifted, no -1)
        const int dc  = tap - (tap / 3) * 3;
        const _Float16* wbase = wt + ((size_t)tap * COUT + cout0) * CIN;
        #pragma unroll
        for (int t = 0; t < 2; ++t) {
            int row = rowbse + t * 64;
            unsigned goff = (unsigned)((row * CIN + cb * BK + kc8) * 2);
            unsigned loff = (unsigned)(size_t)&As[buf][row * LDSS + kc8];
            asm volatile("global_load_async_to_lds_b128 %0, %1, %2"
                         :: "v"(loff), "v"(goff), "s"(wbase) : "memory");
        }
        #pragma unroll
        for (int t = 0; t < 2; ++t) {
            int row = rowbse + t * 64;
            unsigned goff = (unsigned)(((((pb[t] * HP + poh[t] + dr) * WP) +
                                         pow_[t] + dc) * CIN + cb * BK + kc8) * 2);
            unsigned loff = (unsigned)(size_t)&Bs[buf][row * LDSS + kc8];
            asm volatile("global_load_async_to_lds_b128 %0, %1, %2"
                         :: "v"(loff), "v"(goff), "s"(xt) : "memory");
        }
    };

    v8f acc[4][2];
    #pragma unroll
    for (int i = 0; i < 4; ++i)
        #pragma unroll
        for (int j = 0; j < 2; ++j)
            acc[i][j] = (v8f){};

    // prologue: fill buffer 0
    stage(0, 0);
    asm volatile("s_wait_asynccnt 0" ::: "memory");
    __syncthreads();

    for (int s = 0; s < KSTEPS; ++s) {
        const int cur = s & 1;

        // prefetch next K-step into the other buffer (async, no VGPR staging)
        if (s + 1 < KSTEPS) stage(cur ^ 1, s + 1);

        // ---- build fragments per ISA 7.12.2 layouts (2x ds_load_b128 each) ----
        const _Float16* Ac = &As[cur][0];
        const _Float16* Bc = &Bs[cur][0];
        H16 af[4], bf[2];
        #pragma unroll
        for (int mt = 0; mt < 4; ++mt) {
            int rowA = m_wave + mt * 16 + l16;
            int lo   = half * 8;
            af[mt].h[0] = *(const v8h*)&Ac[rowA * LDSS + lo];
            af[mt].h[1] = *(const v8h*)&Ac[rowA * LDSS + lo + 16];
        }
        #pragma unroll
        for (int nt = 0; nt < 2; ++nt) {
            int rowB = n_wave + nt * 16 + l16;
            int klo  = half * 8;
            bf[nt].h[0] = *(const v8h*)&Bc[rowB * LDSS + klo];
            bf[nt].h[1] = *(const v8h*)&Bc[rowB * LDSS + klo + 16];
        }

        // ---- 8x WMMA: D = A*B + C ----
        #pragma unroll
        for (int mt = 0; mt < 4; ++mt)
            #pragma unroll
            for (int nt = 0; nt < 2; ++nt)
                acc[mt][nt] = __builtin_amdgcn_wmma_f32_16x16x32_f16(
                    false, af[mt].v, false, bf[nt].v,
                    (short)0, acc[mt][nt], false, false);

        if (s + 1 < KSTEPS) {
            // next buffer's async copies landed; all waves done reading cur
            asm volatile("s_wait_asynccnt 0" ::: "memory");
            __syncthreads();
        }
    }

    // ---- epilogue: f32 acc -> fp16 NCHW output ----
    #pragma unroll
    for (int nt = 0; nt < 2; ++nt) {
        int p  = pix0 + n_wave + nt * 16 + l16;
        int b  = p / HWSZ;
        int rr = p - b * HWSZ;
        int oh = rr / WW;
        int ow = rr - oh * WW;
        size_t obase = (size_t)b * COUT * HWSZ + (size_t)oh * WW + ow;
        #pragma unroll
        for (int mt = 0; mt < 4; ++mt) {
            int cbase = cout0 + m_wave + mt * 16 + half * 8;
            #pragma unroll
            for (int q = 0; q < 8; ++q)
                out[obase + (size_t)(cbase + q) * HWSZ] = (_Float16)acc[mt][nt][q];
        }
    }
}

// ---------------------------------------------------------------------------
extern "C" void kernel_launch(void* const* d_in, const int* in_sizes, int n_in,
                              void* d_out, int out_size, void* d_ws, size_t ws_size,
                              hipStream_t stream) {
    const _Float16* x = (const _Float16*)d_in[0];
    const float*    w = (const float*)d_in[1];
    _Float16*     out = (_Float16*)d_out;

    _Float16* wt = (_Float16*)d_ws;                      // 9*256*256 fp16 = 1.18 MB
    _Float16* xt = wt + (size_t)9 * COUT * CIN;          // NHWC fp16 + halo, ~213 MB

    nova_wt_xform<<<(9 * COUT * CIN) / 256, 256, 0, stream>>>(w, wt);

    // zero the halo'd NHWC buffer, then fill the interior
    hipMemsetAsync(xt, 0, (size_t)BATCH * HP * WP * CIN * sizeof(_Float16), stream);
    dim3 tgrid((WW + 31) / 32, CIN / 32, BATCH * HH);
    nova_nchw_to_nhwc<<<tgrid, dim3(32, 8), 0, stream>>>(x, xt);

    dim3 ggrid(NPIX / BN, COUT / BM);
    nova_conv_wmma<<<ggrid, 256, 0, stream>>>(xt, wt, out);
}
